// Neural_SDE_integrator_74371653698173
// MI455X (gfx1250) — matrix operand order repaired and checked by reading, not compiled, not run on hardware
//
#include <hip/hip_runtime.h>
#include <cstddef>

typedef __attribute__((ext_vector_type(16))) _Float16 v16h;
typedef __attribute__((ext_vector_type(8)))  _Float16 v8h;
typedef __attribute__((ext_vector_type(8)))  float    v8f;

#define LATENT   100
#define U_DIM    20
#define NSTEP    50
#define NROWS    4096     // BATCH * NUM_SAMPLES
#define DIM_AUG  102
#define T_LEN    51
#define DT       0.05f
// SIGMA * sqrt(DT)
#define G_SQDT   0.11180339887f

// packed f16 weight geometry (element offsets, f16 units)
#define OFF0 0          // L0: Kp=128, Np=400 -> 51200
#define OFF1 51200      // L1..L4: Kp=416, Np=400 -> 166400 each
#define OFF2 217600
#define OFF3 384000
#define OFF4 550400
#define OFF5 716800     // L5: Kp=416, Np=112 -> 46592
#define PW_TOTAL 763392

#define ACT_STRIDE 424  // f16 row stride for LDS activations (16B aligned, conflict-spread)

// hardware tanh if the toolchain exposes it (gfx1250 has V_TANH_F32)
__device__ __forceinline__ float fast_tanhf(float x) {
#if __has_builtin(__builtin_amdgcn_tanhf)
  return __builtin_amdgcn_tanhf(x);
#elif __has_builtin(__builtin_amdgcn_tanh_f32)
  return __builtin_amdgcn_tanh_f32(x);
#else
  return tanhf(x);
#endif
}

// ---------------------------------------------------------------------------
// Kernel 1: treat-MLP table U[50][20] (its input is t_s * ones -> row-invariant)
// ---------------------------------------------------------------------------
__global__ void compute_u_kernel(const float* __restrict__ t,
                                 const float* w0, const float* c0,
                                 const float* w1, const float* c1,
                                 const float* w2, const float* c2,
                                 const float* w3, const float* c3,
                                 const float* w4, const float* c4,
                                 const float* w5, const float* c5,
                                 float* __restrict__ U) {
  int i = threadIdx.x;
  if (i >= NSTEP) return;
  float ts = t[i];
  float h[U_DIM], h2[U_DIM];
  for (int j = 0; j < U_DIM; ++j) h[j] = fmaxf(ts * w0[j] + c0[j], 0.f);
  const float* Ws[4] = {w1, w2, w3, w4};
  const float* Cs[4] = {c1, c2, c3, c4};
  for (int l = 0; l < 4; ++l) {
    for (int j = 0; j < U_DIM; ++j) {
      float s = Cs[l][j];
      for (int k = 0; k < U_DIM; ++k) s += h[k] * Ws[l][k * U_DIM + j];
      h2[j] = fmaxf(s, 0.f);
    }
    for (int j = 0; j < U_DIM; ++j) h[j] = h2[j];
  }
  for (int j = 0; j < U_DIM; ++j) {
    float s = c5[j];
    for (int k = 0; k < U_DIM; ++k) s += h[k] * w5[k * U_DIM + j];
    U[i * U_DIM + j] = s;
  }
}

// ---------------------------------------------------------------------------
// Kernel 2: pack drift weights into f16 WMMA B-fragment layout.
//   n = nt*16 + (lane & 15)
//   k = kt*32 + (lane>=16 ? 8 : 0) + (e < 8 ? e : e + 8)   (e in 0..15)
// stored at pw[ ((nt*numKt + kt)*32 + lane)*16 + e ]
// ---------------------------------------------------------------------------
__global__ void pack_weights_kernel(const float* W0, const float* W1,
                                    const float* W2, const float* W3,
                                    const float* W4, const float* W5,
                                    _Float16* __restrict__ pw) {
  int idx = blockIdx.x * 256 + threadIdx.x;
  if (idx >= PW_TOTAL) return;
  const int offs[7] = {OFF0, OFF1, OFF2, OFF3, OFF4, OFF5, PW_TOTAL};
  const int Ks[6]   = {120, 400, 400, 400, 400, 400};
  const int Ns[6]   = {400, 400, 400, 400, 400, 100};
  const int nKts[6] = {4, 13, 13, 13, 13, 13};
  const float* Wt[6] = {W0, W1, W2, W3, W4, W5};
  int l = 0;
  while (idx >= offs[l + 1]) ++l;
  int rel  = idx - offs[l];
  int tile = rel >> 9;        // 512 elements per fragment tile
  int r    = rel & 511;
  int lane = r >> 4;
  int e    = r & 15;
  int numKt = nKts[l];
  int nt = tile / numKt;
  int kt = tile - nt * numKt;
  int n = nt * 16 + (lane & 15);
  int k = kt * 32 + ((lane & 16) ? 8 : 0) + (e < 8 ? e : e + 8);
  float v = (k < Ks[l] && n < Ns[l]) ? Wt[l][(size_t)k * Ns[l] + n] : 0.f;
  pw[idx] = (_Float16)v;
}

// ---------------------------------------------------------------------------
// Fragment loaders
// ---------------------------------------------------------------------------
__device__ __forceinline__ v16h cat16(v8h lo, v8h hi) {
  return __builtin_shufflevector(lo, hi, 0, 1, 2, 3, 4, 5, 6, 7,
                                 8, 9, 10, 11, 12, 13, 14, 15);
}

// A fragment (16x32 f16) from LDS row-major activations.
__device__ __forceinline__ v16h load_a(const _Float16* src, int lane, int kt) {
  int row = lane & 15;
  int kb  = kt * 32 + ((lane & 16) ? 8 : 0);
  const _Float16* p = src + row * ACT_STRIDE + kb;
  v8h lo = *(const v8h*)p;
  v8h hi = *(const v8h*)(p + 16);
  return cat16(lo, hi);
}

// B fragment: pre-packed, 16 contiguous f16 per lane. nt/kt are scalar.
__device__ __forceinline__ v16h load_b(const _Float16* pw, int lane, int numKt,
                                       int nt, int kt) {
  const _Float16* p = pw + ((size_t)(nt * numKt + kt) * 32 + lane) * 16;
  v8h lo = *(const v8h*)p;
  v8h hi = *(const v8h*)(p + 8);
  return cat16(lo, hi);
}

__device__ __forceinline__ v8f wmma_step(v16h a, v16h b, v8f c) {
  return __builtin_amdgcn_wmma_f32_16x16x32_f16(false, a, false, b, (short)0, c,
                                                false, false);
}

// Bias + activation + f16 store of one 16x16 C tile.
template <int ACT>  // 0 = relu, 1 = tanh
__device__ __forceinline__ void store_tile(_Float16* dst, const float* bias,
                                           v8f acc, int lane, int nt) {
  const int col   = lane & 15;
  const int mbase = (lane & 16) ? 8 : 0;
  const int n     = nt * 16 + col;
  const float bb  = bias[n];
#pragma unroll
  for (int r = 0; r < 8; ++r) {
    float v = acc[r] + bb;
    v = (ACT == 0) ? fmaxf(v, 0.f) : fast_tanhf(v);
    dst[(mbase + r) * ACT_STRIDE + n] = (_Float16)v;
  }
}

// ---------------------------------------------------------------------------
// One MLP layer: M=16, N=NUMNT*16, K=NUMKT*32.  Waves own tiles nt = wave+8*i.
// Main loop is branch-free (every wave owns NFULL tiles); the one leftover
// tile (NUMNT%8) runs in a separate scalar-guarded loop so the hot loop stays
// schedulable (loads batched ahead of WMMAs, no EXEC games).
// ---------------------------------------------------------------------------
template <int ACT, int NUMKT, int NUMNT>
__device__ __forceinline__ void layer_gemm(const _Float16* __restrict__ src,
                                           const _Float16* __restrict__ pw,
                                           const float* __restrict__ bias,
                                           _Float16* __restrict__ dst) {
  const int lane = threadIdx.x & 31;
  const int wave = __builtin_amdgcn_readfirstlane(threadIdx.x) >> 5;  // SGPR
  constexpr int NFULL = NUMNT / 8;
  constexpr int REM   = NUMNT - NFULL * 8;

  if (NFULL > 0) {
    v8f acc[NFULL ? NFULL : 1] = {};
#pragma unroll
    for (int kt = 0; kt < NUMKT; ++kt) {
      v16h a = load_a(src, lane, kt);
#pragma unroll
      for (int i = 0; i < NFULL; ++i) {
        v16h b = load_b(pw, lane, NUMKT, wave + 8 * i, kt);
        acc[i] = wmma_step(a, b, acc[i]);
      }
    }
#pragma unroll
    for (int i = 0; i < NFULL; ++i)
      store_tile<ACT>(dst, bias, acc[i], lane, wave + 8 * i);
  }
  if (REM > 0 && wave < REM) {   // scalar branch: wave is in an SGPR
    const int nt = 8 * NFULL + wave;
    v8f acc = {};
#pragma unroll
    for (int kt = 0; kt < NUMKT; ++kt) {
      v16h a = load_a(src, lane, kt);
      v16h b = load_b(pw, lane, NUMKT, nt, kt);
      acc = wmma_step(a, b, acc);
    }
    store_tile<ACT>(dst, bias, acc, lane, nt);
  }
}

// Last layer: N=100 (7 tiles), linear, f32 result into driftS.
__device__ __forceinline__ void layer_gemm_last(const _Float16* __restrict__ src,
                                                const _Float16* __restrict__ pw,
                                                const float* __restrict__ bias,
                                                float* __restrict__ driftS) {
  const int lane = threadIdx.x & 31;
  const int wave = __builtin_amdgcn_readfirstlane(threadIdx.x) >> 5;
  if (wave < 7) {
    constexpr int NUMKT = 13;
    v8f acc = {};
#pragma unroll
    for (int kt = 0; kt < NUMKT; ++kt) {
      v16h a = load_a(src, lane, kt);
      v16h b = load_b(pw, lane, NUMKT, wave, kt);
      acc = wmma_step(a, b, acc);
    }
    const int n = wave * 16 + (lane & 15);
    if (n < LATENT) {
      const float bb  = bias[n];
      const int mbase = (lane & 16) ? 8 : 0;
#pragma unroll
      for (int r = 0; r < 8; ++r)
        driftS[(mbase + r) * 112 + n] = acc[r] + bb;
    }
  }
}

// ---------------------------------------------------------------------------
// Main kernel: each block owns 16 particles for all 50 Euler-Maruyama steps.
// ---------------------------------------------------------------------------
__global__ __launch_bounds__(256) void sde_kernel(
    const float* __restrict__ z0, const float* __restrict__ Tx,
    const float* __restrict__ noise, const float* __restrict__ Uglob,
    const _Float16* __restrict__ pw,
    const float* b0, const float* b1, const float* b2, const float* b3,
    const float* b4, const float* b5, float* __restrict__ out) {
  __shared__ __attribute__((aligned(16))) _Float16 actA[16 * ACT_STRIDE];
  __shared__ __attribute__((aligned(16))) _Float16 actB[16 * ACT_STRIDE];
  __shared__ float zbuf[16 * 104];
  __shared__ float driftS[16 * 112];
  __shared__ float biasS[2100];
  __shared__ float TxS[16];
  __shared__ float lqS[16];

  const int tid = threadIdx.x;
  const int n0  = blockIdx.x * 16;   // first particle of this block
  float* outLq  = out + (size_t)NROWS * T_LEN * LATENT;

  for (int i = tid; i < 400; i += 256) {
    biasS[i]        = b0[i];
    biasS[400 + i]  = b1[i];
    biasS[800 + i]  = b2[i];
    biasS[1200 + i] = b3[i];
    biasS[1600 + i] = b4[i];
    if (i < 100) biasS[2000 + i] = b5[i];
  }
  if (tid < 16) {
    TxS[tid] = Tx[(n0 + tid) >> 3];  // Tx broadcast over NUM_SAMPLES=8
    lqS[tid] = 0.f;
  }
  for (int idx = tid; idx < 16 * LATENT; idx += 256) {
    int row = idx / LATENT, col = idx - row * LATENT;
    float v = z0[(size_t)(n0 + row) * LATENT + col];
    zbuf[row * 104 + col] = v;
    out[((size_t)(n0 + row) * T_LEN) * LATENT + col] = v;   // t = 0 slice
  }
  // zero K-padding columns 400..423 of both activation buffers
  for (int idx = tid; idx < 16 * 24; idx += 256) {
    int row = idx / 24, c = 400 + (idx - row * 24);
    actA[row * ACT_STRIDE + c] = (_Float16)0.f;
    actB[row * ACT_STRIDE + c] = (_Float16)0.f;
  }
  __syncthreads();

  for (int t = 0; t < NSTEP; ++t) {
    // build x = [z (100), u(t)*Tx (20), zero pad (8)] -> actA cols 0..127
    for (int idx = tid; idx < 16 * 128; idx += 256) {
      int row = idx >> 7, col = idx & 127;
      float v;
      if (col < LATENT)   v = zbuf[row * 104 + col];
      else if (col < 120) v = Uglob[t * U_DIM + (col - LATENT)] * TxS[row];
      else                v = 0.f;
      actA[row * ACT_STRIDE + col] = (_Float16)v;
    }
    __syncthreads();

    layer_gemm<0, 4, 25>(actA, pw + OFF0, biasS + 0,    actB); __syncthreads();
    layer_gemm<1, 13, 25>(actB, pw + OFF1, biasS + 400,  actA); __syncthreads();
    layer_gemm<1, 13, 25>(actA, pw + OFF2, biasS + 800,  actB); __syncthreads();
    layer_gemm<1, 13, 25>(actB, pw + OFF3, biasS + 1200, actA); __syncthreads();
    layer_gemm<1, 13, 25>(actA, pw + OFF4, biasS + 1600, actB); __syncthreads();
    layer_gemm_last(actB, pw + OFF5, biasS + 2000, driftS);     __syncthreads();

    // SDE update: f = drift + 0.1 z ; u_s = 2*drift + 0.4 z
    int row = tid >> 4, sub = tid & 15;
    const float* nrow = noise + ((size_t)t * NROWS + (n0 + row)) * DIM_AUG;
    float* orow = out + ((size_t)(n0 + row) * T_LEN + (t + 1)) * LATENT;
    float partial = 0.f;
    for (int c = sub; c < LATENT; c += 16) {
      float z = zbuf[row * 104 + c];
      float d = driftS[row * 112 + c];
      float us = 2.f * d + 0.4f * z;
      partial += us * us;
      float znew = z + (d + 0.1f * z) * DT + G_SQDT * nrow[c];
      zbuf[row * 104 + c] = znew;
      orow[c] = znew;
    }
    partial += __shfl_xor(partial, 1, 32);
    partial += __shfl_xor(partial, 2, 32);
    partial += __shfl_xor(partial, 4, 32);
    partial += __shfl_xor(partial, 8, 32);
    if (sub == 0) lqS[row] += 0.5f * partial * DT;
    __syncthreads();
  }

  if (tid < 16) outLq[n0 + tid] = lqS[tid];
}

// ---------------------------------------------------------------------------
extern "C" void kernel_launch(void* const* d_in, const int* in_sizes, int n_in,
                              void* d_out, int out_size, void* d_ws,
                              size_t ws_size, hipStream_t stream) {
  (void)in_sizes; (void)n_in; (void)out_size; (void)ws_size;
  const float* z0    = (const float*)d_in[0];
  const float* t     = (const float*)d_in[1];
  const float* Tx    = (const float*)d_in[2];
  const float* noise = (const float*)d_in[3];
  const float* tW[6]; const float* tb[6];
  for (int i = 0; i < 6; ++i) { tW[i] = (const float*)d_in[4 + 2 * i];
                                tb[i] = (const float*)d_in[5 + 2 * i]; }
  const float* dW[6]; const float* db[6];
  for (int i = 0; i < 6; ++i) { dW[i] = (const float*)d_in[16 + 2 * i];
                                db[i] = (const float*)d_in[17 + 2 * i]; }

  float*    U  = (float*)d_ws;                        // 50*20 f32
  _Float16* pw = (_Float16*)((char*)d_ws + 4096);     // packed f16 weights

  compute_u_kernel<<<1, 64, 0, stream>>>(t, tW[0], tb[0], tW[1], tb[1], tW[2],
                                         tb[2], tW[3], tb[3], tW[4], tb[4],
                                         tW[5], tb[5], U);
  pack_weights_kernel<<<(PW_TOTAL + 255) / 256, 256, 0, stream>>>(
      dW[0], dW[1], dW[2], dW[3], dW[4], dW[5], pw);
  sde_kernel<<<NROWS / 16, 256, 0, stream>>>(
      z0, Tx, noise, U, pw, db[0], db[1], db[2], db[3], db[4], db[5],
      (float*)d_out);
}
